// MaskedDeepLSTM_2911987826994
// MI455X (gfx1250) — matrix operand order repaired and checked by reading, not compiled, not run on hardware
//
#include <hip/hip_runtime.h>
#include <math.h>

// Problem constants (match reference)
#define B_ 64
#define T_ 512
#define F_ 1024
#define H_ 1024
#define G_ 4096   // 4*H

typedef __attribute__((ext_vector_type(16))) __bf16 v16bf;
typedef __attribute__((ext_vector_type(8)))  float  v8f;

union Frag32B { v16bf v; uint4 u[2]; };

__device__ __forceinline__ float sigmoidf_(float x) { return 1.0f / (1.0f + __expf(-x)); }

// ---------------------------------------------------------------------------
// Pack masked weight w[G,1024] (f32) into WMMA B-fragment-ordered bf16.
// Layout: [ntile(256)][kchunk(32)][lane(32)][e(16)]  where for the 32x16 (KxN)
// bf16 B fragment: lane%16 = N column, k = 8*(lane/16) + e (+8 for e >= 8) within the chunk.
// Each wave's per-chunk B fragment is then 2 contiguous b128 loads per lane.
// ---------------------------------------------------------------------------
__global__ __launch_bounds__(256) void pack_w_kernel(const float* __restrict__ w,
                                                     const unsigned char* __restrict__ msk,
                                                     __bf16* __restrict__ out) {
  int idx  = blockIdx.x * 256 + threadIdx.x;   // < 4096*1024
  int e    = idx & 15;
  int lane = (idx >> 4) & 31;
  int kc   = (idx >> 9) & 31;
  int nt   = idx >> 14;
  int g    = nt * 16 + (lane & 15);        // output-gate column (N)
  int k    = kc * 32 + ((lane >> 4) << 3) + e + ((e >> 3) << 3);     // reduction index (K)
  size_t src = (size_t)g * 1024 + k;
  float v = w[src] * (msk[src] ? 1.0f : 0.0f);
  out[idx] = (__bf16)v;
}

// bias[g] = b_ih[g] + b_hh[g]
__global__ __launch_bounds__(256) void bias_kernel(const float* __restrict__ bi,
                                                   const float* __restrict__ bh,
                                                   float* __restrict__ out) {
  int idx = blockIdx.x * 256 + threadIdx.x;    // < G_
  out[idx] = bi[idx] + bh[idx];
}

// x [B,T,F] f32 (batch-first) -> xb16 [T,B,F] bf16  ( == [T*B, F] row-major )
__global__ __launch_bounds__(256) void xcast_kernel(const float* __restrict__ x,
                                                    __bf16* __restrict__ out) {
  int idx = blockIdx.x * 256 + threadIdx.x;    // < B_*T_*F_ = 33554432
  int t = idx >> 16;                           // B_*F_ = 65536
  int r = idx & 65535;
  int b = r >> 10;
  int f = r & 1023;
  out[idx] = (__bf16)x[((size_t)b * T_ + t) * F_ + f];
}

__global__ __launch_bounds__(256) void zero_state_kernel(float* __restrict__ c,
                                                         __bf16* __restrict__ h) {
  int idx = blockIdx.x * 256 + threadIdx.x;    // < B_*H_
  c[idx] = 0.0f;
  h[idx] = (__bf16)0.0f;
}

// ---------------------------------------------------------------------------
// Batched input-projection GEMM (off the serial chain):
//   Xp[r, n] = bf16( A[r, :] @ W[:, n] + bias[n] ),  r in [0, T*B)
// Grid: (32, 512) blocks x 256 threads (8 waves). Wave tn = blockIdx.x*8+wave
// owns a 16-wide N stripe; computes 4 M-tiles of the 64-row block r0.
// ---------------------------------------------------------------------------
__global__ __launch_bounds__(256) void proj_gemm(const __bf16* __restrict__ A,   // [T*B, 1024]
                                                 const __bf16* __restrict__ PW,  // packed
                                                 const float* __restrict__ bias, // [4096]
                                                 __bf16* __restrict__ Xp) {      // [T*B, 4096]
  const int lane = threadIdx.x & 31;
  const int tn   = blockIdx.x * 8 + (threadIdx.x >> 5);  // 0..255
  const int r0   = blockIdx.y * 64;                      // row block
  const int rlo  = lane & 15;
  const int koff = (lane >> 4) << 3;

  v8f acc[4] = {};
  const __bf16* Bp = PW + (size_t)tn * (32 * 512) + lane * 16;

  for (int kc = 0; kc < 32; ++kc) {
    Frag32B bf;
    bf.u[0] = *(const uint4*)(Bp + kc * 512);
    bf.u[1] = *(const uint4*)(Bp + kc * 512 + 8);
    const int k0 = kc * 32 + koff;
#pragma unroll
    for (int m = 0; m < 4; ++m) {
      Frag32B af;
      const __bf16* ap = A + ((size_t)(r0 + m * 16 + rlo) << 10) + k0;
      af.u[0] = *(const uint4*)(ap);
      af.u[1] = *(const uint4*)(ap + 16);
      acc[m] = __builtin_amdgcn_wmma_f32_16x16x32_bf16(
          false, af.v, false, bf.v, (short)0, acc[m], false, false);
    }
  }

  const int n   = tn * 16 + rlo;
  const int rhi = (lane >> 4) << 3;
  const float bv = bias[n];
#pragma unroll
  for (int m = 0; m < 4; ++m) {
#pragma unroll
    for (int v = 0; v < 8; ++v) {
      Xp[(size_t)(r0 + m * 16 + v + rhi) * G_ + n] = (__bf16)(acc[m][v] + bv);
    }
  }
}

// ---------------------------------------------------------------------------
// Per-timestep recurrent GEMM (serial critical path, K = 1024 only):
//   gates[64,4096] = f32(Xp_t) + h[64,1024] @ Whh
// Grid: 32 blocks x 256 threads. Each wave: 1 N stripe x 4 M-tiles,
// 32 K-chunks -> 128 WMMAs; B fragments reused 4x across M.
// ---------------------------------------------------------------------------
__global__ __launch_bounds__(256) void rec_gemm(const __bf16* __restrict__ Ah,   // [64,1024]
                                                const __bf16* __restrict__ PWh,  // packed
                                                const __bf16* __restrict__ Xp_t, // [64,4096]
                                                float* __restrict__ gates) {     // [64,4096]
  const int lane = threadIdx.x & 31;
  const int tn   = blockIdx.x * 8 + (threadIdx.x >> 5);  // 0..255
  const int rlo  = lane & 15;
  const int koff = (lane >> 4) << 3;

  v8f acc[4] = {};
  const __bf16* Bp = PWh + (size_t)tn * (32 * 512) + lane * 16;

  for (int kc = 0; kc < 32; ++kc) {
    Frag32B bf;
    bf.u[0] = *(const uint4*)(Bp + kc * 512);
    bf.u[1] = *(const uint4*)(Bp + kc * 512 + 8);
    const int k0 = kc * 32 + koff;
#pragma unroll
    for (int m = 0; m < 4; ++m) {
      Frag32B af;
      const __bf16* ap = Ah + ((size_t)(m * 16 + rlo) << 10) + k0;
      af.u[0] = *(const uint4*)(ap);
      af.u[1] = *(const uint4*)(ap + 16);
      acc[m] = __builtin_amdgcn_wmma_f32_16x16x32_bf16(
          false, af.v, false, bf.v, (short)0, acc[m], false, false);
    }
  }

  const int n   = tn * 16 + rlo;
  const int rhi = (lane >> 4) << 3;
#pragma unroll
  for (int m = 0; m < 4; ++m) {
#pragma unroll
    for (int v = 0; v < 8; ++v) {
      const size_t r = (size_t)(m * 16 + v + rhi) * G_ + n;
      gates[r] = acc[m][v] + (float)Xp_t[r];
    }
  }
}

// ---------------------------------------------------------------------------
// Elementwise LSTM cell: i,f,g,o -> c,h. c kept in f32, h emitted bf16.
// ---------------------------------------------------------------------------
__global__ __launch_bounds__(256) void lstm_cell_kernel(const float* __restrict__ gates,
                                                        float* __restrict__ c,
                                                        __bf16* __restrict__ h,
                                                        __bf16* __restrict__ hseq,
                                                        float* __restrict__ hout) {
  int idx = blockIdx.x * 256 + threadIdx.x;    // < B_*H_
  int b = idx >> 10;
  int j = idx & 1023;
  const float* gr = gates + (size_t)b * G_;
  float ig = sigmoidf_(gr[j]);
  float fg = sigmoidf_(gr[H_ + j]);
  float gg = tanhf(gr[2 * H_ + j]);
  float og = sigmoidf_(gr[3 * H_ + j]);
  float cn = fg * c[idx] + ig * gg;
  c[idx] = cn;
  float hn = og * tanhf(cn);
  __bf16 hb = (__bf16)hn;
  h[idx] = hb;
  if (hseq) hseq[idx] = hb;
  if (hout) hout[idx] = hn;
}

extern "C" void kernel_launch(void* const* d_in, const int* in_sizes, int n_in,
                              void* d_out, int out_size, void* d_ws, size_t ws_size,
                              hipStream_t stream) {
  (void)in_sizes; (void)n_in; (void)out_size; (void)ws_size;

  const float* x    = (const float*)d_in[0];
  const float* wih0 = (const float*)d_in[1];
  const float* whh0 = (const float*)d_in[2];
  const float* bih0 = (const float*)d_in[3];
  const float* bhh0 = (const float*)d_in[4];
  const float* wih1 = (const float*)d_in[5];
  const float* whh1 = (const float*)d_in[6];
  const float* bih1 = (const float*)d_in[7];
  const float* bhh1 = (const float*)d_in[8];
  const unsigned char* mih0 = (const unsigned char*)d_in[9];
  const unsigned char* mhh0 = (const unsigned char*)d_in[10];
  const unsigned char* mih1 = (const unsigned char*)d_in[11];
  const unsigned char* mhh1 = (const unsigned char*)d_in[12];

  // ---- workspace layout (~420 MB), 256B aligned slabs ----
  char* p = (char*)d_ws;
  auto alloc = [&](size_t bytes) -> char* {
    char* r = p;
    p += (bytes + 255) & ~(size_t)255;
    return r;
  };
  const size_t WB = (size_t)G_ * 1024 * sizeof(__bf16);       // 8 MB each
  __bf16* PWih0 = (__bf16*)alloc(WB);
  __bf16* PWhh0 = (__bf16*)alloc(WB);
  __bf16* PWih1 = (__bf16*)alloc(WB);
  __bf16* PWhh1 = (__bf16*)alloc(WB);
  float*  bias0 = (float*)alloc(G_ * sizeof(float));
  float*  bias1 = (float*)alloc(G_ * sizeof(float));
  __bf16* xb16  = (__bf16*)alloc((size_t)T_ * B_ * F_ * sizeof(__bf16));  // 64 MB
  __bf16* h1seq = (__bf16*)alloc((size_t)T_ * B_ * H_ * sizeof(__bf16));  // 64 MB
  __bf16* Xp    = (__bf16*)alloc((size_t)T_ * B_ * G_ * sizeof(__bf16));  // 256 MB (reused per layer)
  float*  gates = (float*)alloc((size_t)B_ * G_ * sizeof(float));          // 1 MB
  float*  cbuf  = (float*)alloc((size_t)B_ * H_ * sizeof(float));
  __bf16* hbuf  = (__bf16*)alloc((size_t)B_ * H_ * sizeof(__bf16));

  // ---- one-time prep ----
  const int packBlocks = (G_ * 1024) / 256;                    // 16384
  pack_w_kernel<<<packBlocks, 256, 0, stream>>>(wih0, mih0, PWih0);
  pack_w_kernel<<<packBlocks, 256, 0, stream>>>(whh0, mhh0, PWhh0);
  pack_w_kernel<<<packBlocks, 256, 0, stream>>>(wih1, mih1, PWih1);
  pack_w_kernel<<<packBlocks, 256, 0, stream>>>(whh1, mhh1, PWhh1);
  bias_kernel<<<G_ / 256, 256, 0, stream>>>(bih0, bhh0, bias0);
  bias_kernel<<<G_ / 256, 256, 0, stream>>>(bih1, bhh1, bias1);
  xcast_kernel<<<(B_ * T_ * F_) / 256, 256, 0, stream>>>(x, xb16);

  const int stateBlocks = (B_ * H_) / 256;                     // 256
  const int recBlocks   = 32;                                  // 256 waves
  const dim3 projGrid(32, (T_ * B_) / 64);                     // (32, 512)

  // ---- layer 0 ----
  // Input projection for all timesteps at once (parallel, off the chain).
  proj_gemm<<<projGrid, 256, 0, stream>>>(xb16, PWih0, bias0, Xp);
  zero_state_kernel<<<stateBlocks, 256, 0, stream>>>(cbuf, hbuf);
  for (int t = 0; t < T_; ++t) {
    rec_gemm<<<recBlocks, 256, 0, stream>>>(
        hbuf, PWhh0, Xp + (size_t)t * B_ * G_, gates);
    lstm_cell_kernel<<<stateBlocks, 256, 0, stream>>>(
        gates, cbuf, hbuf, h1seq + (size_t)t * B_ * H_, nullptr);
  }

  // ---- layer 1 ----
  proj_gemm<<<projGrid, 256, 0, stream>>>(h1seq, PWih1, bias1, Xp);
  zero_state_kernel<<<stateBlocks, 256, 0, stream>>>(cbuf, hbuf);
  for (int t = 0; t < T_; ++t) {
    rec_gemm<<<recBlocks, 256, 0, stream>>>(
        hbuf, PWhh1, Xp + (size_t)t * B_ * G_, gates);
    lstm_cell_kernel<<<stateBlocks, 256, 0, stream>>>(
        gates, cbuf, hbuf, nullptr, (t == T_ - 1) ? (float*)d_out : nullptr);
  }
}
